// TemporalGraphNetworkGAT_19275813224566
// MI455X (gfx1250) — matrix operand order repaired and checked by reading, not compiled, not run on hardware
//
#include <hip/hip_runtime.h>
#include <hip/hip_bf16.h>

typedef __attribute__((ext_vector_type(16))) _Float16 v16h;
typedef __attribute__((ext_vector_type(8)))  float    v8f;

constexpr int Bc = 4, Nc = 1000, Tc = 8, Fc = 64, Hc = 64, HEADSc = 4, Ec = 16000, Pc = 5;
constexpr int Gc  = Bc * Tc;     // 32 graphs
constexpr int E2c = Ec + Nc;     // 17000 edges per graph incl. self-loops
constexpr float NEG_SLOPE = 0.2f;

// =========================================================================
// Batched GEMM via V_WMMA_F32_16X16X32_F16 (K fixed at 64 = 2 WMMA steps).
// C[m,n] = act( sum_k A[m,k] * Bt[n,k] + bias[n] )   (B passed TRANSPOSED)
// One wave computes a 16 x (16*NT) strip: A fragment loaded once per K-step
// and reused across NT B tiles (NT WMMAs back to back). OOB rows/cols are
// clamped (they only feed OOB outputs, never stored), so all loads are
// unconditional 16B-aligned b128 loads. Layouts per ISA 7.12.2 (wave32).
// =========================================================================
__device__ inline v16h load_frag_a(const float* __restrict__ arow, int k0, int lhi)
{
  // halves: j<8 -> k = k0 + lhi*8 + j ; j>=8 -> k = k0 + 16 + lhi*8 + (j-8)
  const float4* pa0 = (const float4*)(arow + k0 + (lhi << 3));
  const float4* pa2 = (const float4*)(arow + k0 + 16 + (lhi << 3));
  float4 a0 = pa0[0], a1 = pa0[1], a2 = pa2[0], a3 = pa2[1];
  v16h a;
  a[0]  = (_Float16)a0.x; a[1]  = (_Float16)a0.y; a[2]  = (_Float16)a0.z; a[3]  = (_Float16)a0.w;
  a[4]  = (_Float16)a1.x; a[5]  = (_Float16)a1.y; a[6]  = (_Float16)a1.z; a[7]  = (_Float16)a1.w;
  a[8]  = (_Float16)a2.x; a[9]  = (_Float16)a2.y; a[10] = (_Float16)a2.z; a[11] = (_Float16)a2.w;
  a[12] = (_Float16)a3.x; a[13] = (_Float16)a3.y; a[14] = (_Float16)a3.z; a[15] = (_Float16)a3.w;
  return a;
}

__device__ inline v16h load_frag_b(const float* __restrict__ brow, int k0, int lhi)
{
  // halves: j<8 -> k = k0 + lhi*8 + j ; j>=8 -> k = k0 + 16 + lhi*8 + (j-8)
  const float4* pb0 = (const float4*)(brow + k0 + (lhi << 3));
  const float4* pb2 = (const float4*)(brow + k0 + 16 + (lhi << 3));
  float4 b0 = pb0[0], b1 = pb0[1], b2 = pb2[0], b3 = pb2[1];
  v16h b;
  b[0]  = (_Float16)b0.x; b[1]  = (_Float16)b0.y; b[2]  = (_Float16)b0.z; b[3]  = (_Float16)b0.w;
  b[4]  = (_Float16)b1.x; b[5]  = (_Float16)b1.y; b[6]  = (_Float16)b1.z; b[7]  = (_Float16)b1.w;
  b[8]  = (_Float16)b2.x; b[9]  = (_Float16)b2.y; b[10] = (_Float16)b2.z; b[11] = (_Float16)b2.w;
  b[12] = (_Float16)b3.x; b[13] = (_Float16)b3.y; b[14] = (_Float16)b3.z; b[15] = (_Float16)b3.w;
  return b;
}

template<int NT>
__global__ void wmma_gemm_kernel(const float* __restrict__ A, long long sA,
                                 const float* __restrict__ Bt, long long sB,
                                 const float* __restrict__ bias,
                                 float* __restrict__ C, long long sC,
                                 int M, int N,
                                 int lda, int ldb, int ldc, int act)
{
  const int batch = blockIdx.y;
  const float* Ab = A + (long long)batch * sA;
  const float* Bb = Bt + (long long)batch * sB;
  float* Cb = C + (long long)batch * sC;

  const int TW  = 16 * NT;
  const int ntg = (N + TW - 1) / TW;
  const int mt  = blockIdx.x / ntg;
  const int ng  = blockIdx.x % ntg;
  const int m0  = mt << 4;
  const int n0  = ng * TW;
  const int lane = threadIdx.x & 31;
  const int l16  = lane & 15;
  const int lhi  = lane >> 4;

  const int mA = min(m0 + l16, M - 1);   // clamp: OOB rows feed only OOB C rows
  const float* arow = Ab + (long long)mA * lda;
  const float* brow[NT];
  #pragma unroll
  for (int j = 0; j < NT; ++j) {
    const int nB = min(n0 + j * 16 + l16, N - 1);
    brow[j] = Bb + (long long)nB * ldb;
  }

  v8f acc[NT];
  #pragma unroll
  for (int j = 0; j < NT; ++j) { v8f z = {}; acc[j] = z; }

  #pragma unroll
  for (int ks = 0; ks < 2; ++ks) {
    const int k0 = ks * 32;
    const v16h a = load_frag_a(arow, k0, lhi);
    #pragma unroll
    for (int j = 0; j < NT; ++j) {
      const v16h b = load_frag_b(brow[j], k0, lhi);
      acc[j] = __builtin_amdgcn_wmma_f32_16x16x32_f16(false, a, false, b,
                                                      (short)0, acc[j], false, false);
    }
  }

  // epilogue: fast unguarded path for interior tiles (wave-uniform check)
  if (m0 + 16 <= M && n0 + TW <= N) {
    #pragma unroll
    for (int j = 0; j < NT; ++j) {
      const int nS = n0 + j * 16 + l16;
      const float bv = bias ? bias[nS] : 0.0f;
      #pragma unroll
      for (int i = 0; i < 8; ++i) {
        const int m = m0 + i + (lhi << 3);
        float v = acc[j][i] + bv;
        if (act) v = v > 0.0f ? v : 0.0f;
        Cb[(long long)m * ldc + nS] = v;
      }
    }
  } else {
    #pragma unroll
    for (int j = 0; j < NT; ++j) {
      const int nS = n0 + j * 16 + l16;
      if (nS < N) {
        const float bv = bias ? bias[nS] : 0.0f;
        #pragma unroll
        for (int i = 0; i < 8; ++i) {
          const int m = m0 + i + (lhi << 3);
          if (m < M) {
            float v = acc[j][i] + bv;
            if (act) v = v > 0.0f ? v : 0.0f;
            Cb[(long long)m * ldc + nS] = v;
          }
        }
      }
    }
  }
}

// ---------- helpers ----------
__global__ void fill_kernel(float* __restrict__ p, float v, long long n) {
  long long i = (long long)blockIdx.x * blockDim.x + threadIdx.x;
  if (i < n) p[i] = v;
}

// W [K,N] -> Wt [N,K]
__global__ void transpose_kernel(const float* __restrict__ W, float* __restrict__ Wt,
                                 int K, int N)
{
  long long idx = (long long)blockIdx.x * blockDim.x + threadIdx.x;
  if (idx >= (long long)K * N) return;
  int k = (int)(idx / N), n = (int)(idx % N);
  Wt[(long long)n * K + k] = W[idx];
}

// x [B,N,T,F] -> xg [G=B*T, N, F]
__global__ void pack_x_kernel(const float* __restrict__ x, float* __restrict__ xg) {
  long long idx = (long long)blockIdx.x * blockDim.x + threadIdx.x;
  const long long total = (long long)Gc * Nc * Fc;
  if (idx >= total) return;
  int f = (int)(idx & 63);
  int n = (int)((idx >> 6) % Nc);
  int g = (int)(idx / ((long long)Fc * Nc));
  int b = g / Tc, t = g % Tc;
  xg[idx] = x[(((long long)b * Nc + n) * Tc + t) * Fc + f];
}

// asrc[g,n,h] = <Hproj[g,n,h,:], att_src[h,:]>  (and adst likewise)
__global__ void attn_score_kernel(const float* __restrict__ Hproj,
                                  const float* __restrict__ att_src,
                                  const float* __restrict__ att_dst,
                                  float* __restrict__ asrc,
                                  float* __restrict__ adst)
{
  long long idx = (long long)blockIdx.x * blockDim.x + threadIdx.x;
  const long long total = (long long)Gc * Nc * HEADSc;
  if (idx >= total) return;
  int h  = (int)(idx & 3);
  long long gn = idx >> 2;
  const float4* hp = (const float4*)(Hproj + gn * (HEADSc * Hc) + h * Hc);
  const float4* vs = (const float4*)(att_src + h * Hc);
  const float4* vd = (const float4*)(att_dst + h * Hc);
  float s1 = 0.0f, s2 = 0.0f;
  #pragma unroll
  for (int k = 0; k < Hc / 4; ++k) {
    float4 xv = hp[k], sv = vs[k], dv = vd[k];
    s1 += xv.x * sv.x + xv.y * sv.y + xv.z * sv.z + xv.w * sv.w;
    s2 += xv.x * dv.x + xv.y * dv.y + xv.z * dv.z + xv.w * dv.w;
  }
  asrc[idx] = s1;
  adst[idx] = s2;
}

__device__ inline void atomicMaxFloat(float* addr, float val) {
  unsigned int* ai = reinterpret_cast<unsigned int*>(addr);
  unsigned int old = *ai;
  while (__uint_as_float(old) < val) {
    unsigned int assumed = old;
    old = atomicCAS(ai, assumed, __float_as_uint(val));
    if (old == assumed) break;
  }
}

__device__ inline void decode_edge(long long idx, int& g, int& e, int& h,
                                   const int* __restrict__ ei, int& src, int& dst)
{
  h = (int)(idx & 3);
  long long ge = idx >> 2;
  e = (int)(ge % E2c);
  g = (int)(ge / E2c);
  if (e < Ec) {
    const int* eb = ei + (long long)g * 2 * Ec;   // [2,E] block for graph g
    src = eb[e];
    dst = eb[Ec + e];
    src = min(max(src, 0), Nc - 1);
    dst = min(max(dst, 0), Nc - 1);
  } else {
    src = dst = e - Ec;                            // appended self-loop
  }
}

__global__ void edge_max_kernel(const int* __restrict__ ei,
                                const float* __restrict__ asrc,
                                const float* __restrict__ adst,
                                float* __restrict__ mbuf)
{
  long long idx = (long long)blockIdx.x * blockDim.x + threadIdx.x;
  const long long total = (long long)Gc * E2c * HEADSc;
  if (idx >= total) return;
  int g, e, h, src, dst;
  decode_edge(idx, g, e, h, ei, src, dst);
  float v = asrc[((long long)g * Nc + src) * HEADSc + h] +
            adst[((long long)g * Nc + dst) * HEADSc + h];
  v = v > 0.0f ? v : NEG_SLOPE * v;
  atomicMaxFloat(&mbuf[((long long)g * Nc + dst) * HEADSc + h], v);
}

__global__ void edge_expsum_kernel(const int* __restrict__ ei,
                                   const float* __restrict__ asrc,
                                   const float* __restrict__ adst,
                                   const float* __restrict__ mbuf,
                                   float* __restrict__ ebuf,
                                   float* __restrict__ sbuf)
{
  long long idx = (long long)blockIdx.x * blockDim.x + threadIdx.x;
  const long long total = (long long)Gc * E2c * HEADSc;
  if (idx >= total) return;
  int g, e, h, src, dst;
  decode_edge(idx, g, e, h, ei, src, dst);
  float v = asrc[((long long)g * Nc + src) * HEADSc + h] +
            adst[((long long)g * Nc + dst) * HEADSc + h];
  v = v > 0.0f ? v : NEG_SLOPE * v;
  float ex = __expf(v - mbuf[((long long)g * Nc + dst) * HEADSc + h]);
  ebuf[idx] = ex;
  atomicAdd(&sbuf[((long long)g * Nc + dst) * HEADSc + h], ex);
}

__global__ void edge_agg_kernel(const int* __restrict__ ei,
                                const float* __restrict__ ebuf,
                                const float* __restrict__ sbuf,
                                const float* __restrict__ Hproj,
                                float* __restrict__ agg)
{
  long long idx = (long long)blockIdx.x * blockDim.x + threadIdx.x;
  const long long total = (long long)Gc * E2c * HEADSc;
  if (idx >= total) return;
  int g, e, h, src, dst;
  decode_edge(idx, g, e, h, ei, src, dst);
  float alpha = ebuf[idx] / sbuf[((long long)g * Nc + dst) * HEADSc + h];
  const float* hs = Hproj + ((long long)g * Nc + src) * (HEADSc * Hc) + h * Hc;
  float* ao       = agg   + ((long long)g * Nc + dst) * (HEADSc * Hc) + h * Hc;
  #pragma unroll 4
  for (int k = 0; k < Hc; ++k) atomicAdd(&ao[k], alpha * hs[k]);
}

// out[g,n,k] = act( mean_h agg[g,n,h,k] + bias[k] )
__global__ void gat_finalize_kernel(const float* __restrict__ agg,
                                    const float* __restrict__ bias,
                                    float* __restrict__ out, int act)
{
  long long idx = (long long)blockIdx.x * blockDim.x + threadIdx.x;
  const long long total = (long long)Gc * Nc * Hc;
  if (idx >= total) return;
  int k = (int)(idx & 63);
  long long gn = idx >> 6;
  const float* a = agg + gn * (HEADSc * Hc);
  float v = 0.25f * (a[k] + a[Hc + k] + a[2 * Hc + k] + a[3 * Hc + k]) + bias[k];
  if (act) v = v > 0.0f ? v : 0.0f;
  out[idx] = v;
}

// fused GRU gate math: h = (1-z)*tanh(i_n + r*h_n) + z*h
__global__ void gru_gate_kernel(const float* __restrict__ gi,
                                const float* __restrict__ gh,
                                const float* __restrict__ bih,
                                const float* __restrict__ bhh,
                                float* __restrict__ h, int M)
{
  long long idx = (long long)blockIdx.x * blockDim.x + threadIdx.x;
  if (idx >= (long long)M * Hc) return;
  int r = (int)(idx >> 6), c = (int)(idx & 63);
  const float* gir = gi + (long long)r * 3 * Hc;
  const float* ghr = gh + (long long)r * 3 * Hc;
  float i_r = gir[c]          + bih[c];
  float i_z = gir[Hc + c]     + bih[Hc + c];
  float i_n = gir[2 * Hc + c] + bih[2 * Hc + c];
  float h_r = ghr[c]          + bhh[c];
  float h_z = ghr[Hc + c]     + bhh[Hc + c];
  float h_n = ghr[2 * Hc + c] + bhh[2 * Hc + c];
  float rr = 1.0f / (1.0f + __expf(-(i_r + h_r)));
  float zz = 1.0f / (1.0f + __expf(-(i_z + h_z)));
  float nn = tanhf(i_n + rr * h_n);
  h[idx] = (1.0f - zz) * nn + zz * h[idx];
}

// =========================================================================
extern "C" void kernel_launch(void* const* d_in, const int* in_sizes, int n_in,
                              void* d_out, int out_size, void* d_ws, size_t ws_size,
                              hipStream_t stream)
{
  (void)in_sizes; (void)n_in; (void)out_size;
  const float* x    = (const float*)d_in[0];
  const int*   ei   = (const int*)d_in[1];
  // d_in[2] edge_feat_seq, d_in[11] edge_W, d_in[12] edge_b: dead code in reference
  const float* W1   = (const float*)d_in[3];
  const float* as1  = (const float*)d_in[4];
  const float* ad1  = (const float*)d_in[5];
  const float* b1   = (const float*)d_in[6];
  const float* W2   = (const float*)d_in[7];
  const float* as2  = (const float*)d_in[8];
  const float* ad2  = (const float*)d_in[9];
  const float* b2   = (const float*)d_in[10];
  const float* Wih0 = (const float*)d_in[13];
  const float* Whh0 = (const float*)d_in[14];
  const float* bih0 = (const float*)d_in[15];
  const float* bhh0 = (const float*)d_in[16];
  const float* Wih1 = (const float*)d_in[17];
  const float* Whh1 = (const float*)d_in[18];
  const float* bih1 = (const float*)d_in[19];
  const float* bhh1 = (const float*)d_in[20];
  const float* pW1  = (const float*)d_in[21];
  const float* pb1  = (const float*)d_in[22];
  const float* pW2  = (const float*)d_in[23];
  const float* pb2  = (const float*)d_in[24];

  // ---- workspace layout (floats), all L2-resident (~95 MB << 192 MB L2) ----
  float* ws = (float*)d_ws;
  size_t o = 0;
  float* Hproj = ws + o; o += (size_t)Gc * Nc * HEADSc * Hc;   // 8,192,000
  float* agg   = ws + o; o += (size_t)Gc * Nc * HEADSc * Hc;   // 8,192,000
  float* ebuf  = ws + o; o += (size_t)Gc * E2c * HEADSc;       // 2,176,000 (reused by GRU)
  float* xg    = ws + o; o += (size_t)Gc * Nc * Hc;            // 2,048,000 (reused as feat1)
  float* feat2 = ws + o; o += (size_t)Gc * Nc * Hc;            // 2,048,000
  float* asrcB = ws + o; o += (size_t)Gc * Nc * HEADSc;
  float* adstB = ws + o; o += (size_t)Gc * Nc * HEADSc;
  float* mbuf  = ws + o; o += (size_t)Gc * Nc * HEADSc;
  float* sbuf  = ws + o; o += (size_t)Gc * Nc * HEADSc;
  float* h1s   = ws + o; o += (size_t)Bc * Nc * Hc;
  float* h2s   = ws + o; o += (size_t)Bc * Nc * Hc;
  float* Wt1   = ws + o; o += (size_t)Fc * HEADSc * Hc;        // 16384 ([256,64])
  float* Wt2   = ws + o; o += (size_t)Hc * HEADSc * Hc;        // 16384
  float* pW1t  = ws + o; o += (size_t)Hc * Hc;                 // 4096  ([64,64])
  float* pW2t  = ws + o; o += (size_t)2 * Pc * Hc;             // 640   ([10,64])
  if (ws_size < o * sizeof(float)) return;   // not enough scratch
  float* gi  = ebuf;                          // [4000,192]
  float* gh  = ebuf + (size_t)Bc * Nc * 3 * Hc;
  float* mid = gh   + (size_t)Bc * Nc * 3 * Hc;

  auto fill = [&](float* p, float v, long long n) {
    fill_kernel<<<(unsigned)((n + 255) / 256), 256, 0, stream>>>(p, v, n);
  };
  auto gemm = [&](const float* A, long long sA, const float* Bt, long long sB,
                  const float* bias, float* C, long long sC,
                  int M, int Nn, int lda, int ldb, int ldc,
                  int act, int batches) {
    const int mt = (M + 15) / 16;
    if (Nn % 64 == 0) {
      dim3 grid((unsigned)(mt * (Nn / 64)), (unsigned)batches);
      wmma_gemm_kernel<4><<<grid, 32, 0, stream>>>(A, sA, Bt, sB, bias, C, sC,
                                                   M, Nn, lda, ldb, ldc, act);
    } else {
      dim3 grid((unsigned)(mt * ((Nn + 15) / 16)), (unsigned)batches);
      wmma_gemm_kernel<1><<<grid, 32, 0, stream>>>(A, sA, Bt, sB, bias, C, sC,
                                                   M, Nn, lda, ldb, ldc, act);
    }
  };

  // ---- one-time weight transposes to [N,K] and x pack ----
  transpose_kernel<<<(Fc * HEADSc * Hc + 255) / 256, 256, 0, stream>>>(W1, Wt1, Fc, HEADSc * Hc);
  transpose_kernel<<<(Hc * HEADSc * Hc + 255) / 256, 256, 0, stream>>>(W2, Wt2, Hc, HEADSc * Hc);
  transpose_kernel<<<(Hc * Hc + 255) / 256, 256, 0, stream>>>(pW1, pW1t, Hc, Hc);
  transpose_kernel<<<(Hc * 2 * Pc + 255) / 256, 256, 0, stream>>>(pW2, pW2t, Hc, 2 * Pc);
  {
    long long n = (long long)Gc * Nc * Fc;
    pack_x_kernel<<<(unsigned)((n + 255) / 256), 256, 0, stream>>>(x, xg);
  }

  // ---- one GAT layer ----
  auto run_gat = [&](const float* inF, const float* Wt, const float* avs,
                     const float* avd, const float* bias, float* outF, int act) {
    // projection: [N,64] @ [64,256] per graph (WMMA, B pre-transposed)
    gemm(inF, (long long)Nc * Hc, Wt, 0, nullptr, Hproj, (long long)Nc * HEADSc * Hc,
         Nc, HEADSc * Hc, Hc, Hc, HEADSc * Hc, 0, Gc);
    long long nn = (long long)Gc * Nc * HEADSc;
    attn_score_kernel<<<(unsigned)((nn + 255) / 256), 256, 0, stream>>>(Hproj, avs, avd, asrcB, adstB);
    fill(mbuf, -3.0e38f, nn);
    fill(sbuf, 0.0f, nn);
    fill(agg, 0.0f, (long long)Gc * Nc * HEADSc * Hc);
    long long ne = (long long)Gc * E2c * HEADSc;
    unsigned eb = (unsigned)((ne + 255) / 256);
    edge_max_kernel<<<eb, 256, 0, stream>>>(ei, asrcB, adstB, mbuf);
    edge_expsum_kernel<<<eb, 256, 0, stream>>>(ei, asrcB, adstB, mbuf, ebuf, sbuf);
    edge_agg_kernel<<<eb, 256, 0, stream>>>(ei, ebuf, sbuf, Hproj, agg);
    long long nf = (long long)Gc * Nc * Hc;
    gat_finalize_kernel<<<(unsigned)((nf + 255) / 256), 256, 0, stream>>>(agg, bias, outF, act);
  };

  run_gat(xg, Wt1, as1, ad1, b1, xg,    1);   // layer 1 (ReLU), output reuses xg
  run_gat(xg, Wt2, as2, ad2, b2, feat2, 0);   // layer 2

  // ---- 2-layer GRU over T=8 ----
  fill(h1s, 0.0f, (long long)Bc * Nc * Hc);
  fill(h2s, 0.0f, (long long)Bc * Nc * Hc);
  const int M4 = Bc * Nc; // 4000
  for (int t = 0; t < Tc; ++t) {
    // layer 0: x_t rows are (b,n); per-batch base feat2 + (b*T + t)*N*H
    // Wih/Whh are [3H,H] = [N,K] already (gi = x @ Wih^T)
    gemm(feat2 + (long long)t * Nc * Hc, (long long)Tc * Nc * Hc, Wih0, 0, nullptr,
         gi, (long long)Nc * 3 * Hc, Nc, 3 * Hc, Hc, Hc, 3 * Hc, 0, Bc);
    gemm(h1s, 0, Whh0, 0, nullptr, gh, 0, M4, 3 * Hc, Hc, Hc, 3 * Hc, 0, 1);
    gru_gate_kernel<<<(M4 * Hc + 255) / 256, 256, 0, stream>>>(gi, gh, bih0, bhh0, h1s, M4);
    // layer 1
    gemm(h1s, 0, Wih1, 0, nullptr, gi, 0, M4, 3 * Hc, Hc, Hc, 3 * Hc, 0, 1);
    gemm(h2s, 0, Whh1, 0, nullptr, gh, 0, M4, 3 * Hc, Hc, Hc, 3 * Hc, 0, 1);
    gru_gate_kernel<<<(M4 * Hc + 255) / 256, 256, 0, stream>>>(gi, gh, bih1, bhh1, h2s, M4);
  }

  // ---- MLP head: mid = relu(hT @ pW1 + pb1); pred = mid @ pW2 + pb2 -> d_out ----
  gemm(h2s, 0, pW1t, 0, pb1, mid, 0, M4, Hc, Hc, Hc, Hc, 1, 1);
  gemm(mid, 0, pW2t, 0, pb2, (float*)d_out, 0, M4, 2 * Pc, Hc, Hc, 2 * Pc, 0, 1);
}
